// EngramMemory_17910013624482
// MI455X (gfx1250) — matrix operand order repaired and checked by reading, not compiled, not run on hardware
//
#include <hip/hip_runtime.h>
#include <hip/hip_bf16.h>
#include <cstdint>

typedef __attribute__((ext_vector_type(16))) __bf16 v16bf;
typedef __attribute__((ext_vector_type(8)))  float  v8f;

#define N_TOK   8192
#define HDIM    2048
#define MDIM    2048
#define SLOTS   8
#define SLOT_DIM 256
#define BUCKETS 100000

static __device__ __forceinline__ unsigned short f32_to_bf16(float f) {
  union { float f; uint32_t u; } c; c.f = f;
  uint32_t u = c.u;
  uint32_t r = u + 0x7FFFu + ((u >> 16) & 1u);   // round-to-nearest-even
  return (unsigned short)(r >> 16);
}

static __device__ __forceinline__ v16bf toV16(uint4 a, uint4 b) {
  union U { uint4 q[2]; v16bf v; } u;
  u.q[0] = a; u.q[1] = b;
  return u.v;
}

// low 32 bits of a generic pointer to __shared__ = LDS byte offset
static __device__ __forceinline__ unsigned ldsOff(const void* p) {
  return (unsigned)(uintptr_t)p;
}

// gfx1250 async global -> LDS, 16B per lane, tracked with ASYNCcnt
static __device__ __forceinline__ void asyncLoadB128(unsigned lds, const void* g) {
  asm volatile("global_load_async_to_lds_b128 %0, %1, off"
               :: "v"(lds), "v"((uint64_t)(uintptr_t)g) : "memory");
}
static __device__ __forceinline__ void waitAsync0() {
  asm volatile("s_wait_asynccnt 0x0" ::: "memory");
}

// ---------------------------------------------------------------------------
// Kernel 1: multi-table gather -> bf16 memory matrix [N_TOK, MDIM]
// ---------------------------------------------------------------------------
__global__ __launch_bounds__(256) void gather_bf16_kernel(
    const float* __restrict__ tables,        // [SLOTS, BUCKETS, SLOT_DIM]
    const int*   __restrict__ ids,           // [N_TOK, SLOTS]
    unsigned short* __restrict__ memB) {     // [N_TOK, MDIM] bf16
  const int n = blockIdx.x;
  const int d = threadIdx.x;
  const int* row = ids + n * SLOTS;
  unsigned short* dst = memB + (size_t)n * MDIM;
#pragma unroll
  for (int s = 0; s < SLOTS; ++s) {
    const int id = row[s];
    const float* src = tables + ((size_t)s * BUCKETS + (size_t)id) * SLOT_DIM;
    dst[s * SLOT_DIM + d] = f32_to_bf16(src[d]);
  }
}

// ---------------------------------------------------------------------------
// Kernel 2: convert key_w / value_w (f32 [HDIM, MDIM]) -> bf16
// ---------------------------------------------------------------------------
__global__ __launch_bounds__(256) void cvt_w_kernel(
    const float* __restrict__ kw, const float* __restrict__ vw,
    unsigned short* __restrict__ kB, unsigned short* __restrict__ vB) {
  const size_t i = ((size_t)blockIdx.x * 256 + threadIdx.x) * 4;
  float4 a = *(const float4*)(kw + i);
  kB[i + 0] = f32_to_bf16(a.x);
  kB[i + 1] = f32_to_bf16(a.y);
  kB[i + 2] = f32_to_bf16(a.z);
  kB[i + 3] = f32_to_bf16(a.w);
  float4 b = *(const float4*)(vw + i);
  vB[i + 0] = f32_to_bf16(b.x);
  vB[i + 1] = f32_to_bf16(b.y);
  vB[i + 2] = f32_to_bf16(b.z);
  vB[i + 3] = f32_to_bf16(b.w);
}

// ---------------------------------------------------------------------------
// Kernel 3: dual GEMM  Ck = mem @ key_w.T , Cv = mem @ value_w.T  (bf16 WMMA)
// Block tile: 128 rows x 64 cols, BOTH outputs. 8 waves:
//   wm = wid&3 -> 32-row slice (2 WMMA row-tiles),  wo = wid>>2 -> {key,value}
// K tile = 64, double-buffered LDS staged by GLOBAL_LOAD_ASYNC_TO_LDS_B128
// (ASYNCcnt): tile t+1 streams into LDS while 16 WMMAs consume tile t.
// Per K-substep: 2 A frags + 4 B frags -> 8 back-to-back v_wmma.
// ---------------------------------------------------------------------------
__global__ __launch_bounds__(256) void dual_gemm_bf16_kernel(
    const unsigned short* __restrict__ A,    // [N_TOK, MDIM] bf16
    const unsigned short* __restrict__ Bk,   // [HDIM, MDIM] bf16 (h-major)
    const unsigned short* __restrict__ Bv,   // [HDIM, MDIM] bf16
    float* __restrict__ Ck,                  // [N_TOK, HDIM]
    float* __restrict__ Cv) {                // [N_TOK, HDIM]
  __shared__ __align__(16) unsigned short lA [2][128 * 64];  // 32 KB
  __shared__ __align__(16) unsigned short lBk[2][ 64 * 64];  // 16 KB
  __shared__ __align__(16) unsigned short lBv[2][ 64 * 64];  // 16 KB

  const int tid  = threadIdx.x;
  const int lane = tid & 31;
  const int wid  = tid >> 5;
  const int wm   = wid & 3;     // 32-row slice within 128-row tile
  const int wo   = wid >> 2;    // 0 = key GEMM, 1 = value GEMM
  const int nBase = blockIdx.y * 128;
  const int hBase = blockIdx.x * 64;

  const int g = lane >> 4;      // half-wave selector
  const int r = lane & 15;

  // staging geometry: 16B chunks; thread owns (srow + 32*j, sch)
  const int srow = tid >> 3;          // 0..31
  const int sch  = (tid & 7) * 8;     // 0..56 halves
  const unsigned short* gA  = A  + (size_t)(nBase + srow) * MDIM + sch;
  const unsigned short* gBk = Bk + (size_t)(hBase + srow) * MDIM + sch;
  const unsigned short* gBv = Bv + (size_t)(hBase + srow) * MDIM + sch;
  const size_t rowSkip = (size_t)32 * MDIM;   // +32 rows per extra chunk

  v8f acc[2][4];
#pragma unroll
  for (int rt = 0; rt < 2; ++rt)
#pragma unroll
    for (int t = 0; t < 4; ++t)
      acc[rt][t] = (v8f){0.f,0.f,0.f,0.f,0.f,0.f,0.f,0.f};

  auto stage = [&](int buf, int k0) {
    const int lbase = srow * 64 + sch;
#pragma unroll
    for (int j = 0; j < 4; ++j)      // A: 128 rows
      asyncLoadB128(ldsOff(&lA[buf][lbase + j * (32 * 64)]),
                    gA + k0 + (size_t)j * rowSkip);
#pragma unroll
    for (int j = 0; j < 2; ++j) {    // B: 64 rows each
      asyncLoadB128(ldsOff(&lBk[buf][lbase + j * (32 * 64)]),
                    gBk + k0 + (size_t)j * rowSkip);
      asyncLoadB128(ldsOff(&lBv[buf][lbase + j * (32 * 64)]),
                    gBv + k0 + (size_t)j * rowSkip);
    }
  };

  stage(0, 0);
  waitAsync0();
  __syncthreads();

  for (int k0 = 0; k0 < MDIM; k0 += 64) {
    const int cur = (k0 >> 6) & 1;
    if (k0 + 64 < MDIM) stage(cur ^ 1, k0 + 64);   // async prefetch next tile

    const unsigned short* lAb = &lA[cur][0];
    const unsigned short* lBb = wo ? &lBv[cur][0] : &lBk[cur][0];
#pragma unroll
    for (int ks = 0; ks < 64; ks += 32) {
      // A fragments (16x32 bf16): lane r, half-wave g -> K chunks {8g, 16+8g}
      v16bf af[2];
#pragma unroll
      for (int rt = 0; rt < 2; ++rt) {
        const int arow = (32 * wm + 16 * rt + r) * 64 + ks;
        uint4 a0 = *(const uint4*)&lAb[arow + 8 * g];
        uint4 a1 = *(const uint4*)&lAb[arow + 16 + 8 * g];
        af[rt] = toV16(a0, a1);
      }
      // B fragments (32x16): column = 16*nt + r, K chunks {8g, 16+8g}
      v16bf bf[4];
#pragma unroll
      for (int nt = 0; nt < 4; ++nt) {
        const int brow = (16 * nt + r) * 64 + ks;
        uint4 b0 = *(const uint4*)&lBb[brow + 8 * g];
        uint4 b1 = *(const uint4*)&lBb[brow + 16 + 8 * g];
        bf[nt] = toV16(b0, b1);
      }
      // 8 back-to-back WMMAs
#pragma unroll
      for (int rt = 0; rt < 2; ++rt)
#pragma unroll
        for (int nt = 0; nt < 4; ++nt)
          acc[rt][nt] = __builtin_amdgcn_wmma_f32_16x16x32_bf16(
              false, af[rt], false, bf[nt], (short)0, acc[rt][nt], false, false);
    }

    waitAsync0();      // next tile fully resident in LDS
    __syncthreads();   // all waves done reading cur + done staging
  }

  float* C = wo ? Cv : Ck;
#pragma unroll
  for (int rt = 0; rt < 2; ++rt)
#pragma unroll
    for (int nt = 0; nt < 4; ++nt)
#pragma unroll
      for (int i = 0; i < 8; ++i) {
        const int row = nBase + 32 * wm + 16 * rt + i + 8 * g;
        const int col = hBase + 16 * nt + r;
        C[(size_t)row * HDIM + col] = acc[rt][nt][i];
      }
}

// ---------------------------------------------------------------------------
// Kernel 4: fused epilogue per token row
// ---------------------------------------------------------------------------
__global__ __launch_bounds__(256) void epilogue_kernel(
    const float* __restrict__ K, const float* __restrict__ V,
    const float* __restrict__ hidden,
    const float* __restrict__ qn, const float* __restrict__ kn,
    const float* __restrict__ vn,
    const float* __restrict__ cw, const float* __restrict__ cb,
    float* __restrict__ out) {
  __shared__ float red[8 * 4];
  const int n   = blockIdx.x;
  const int tid = threadIdx.x;
  const int lane = tid & 31;
  const int wid  = tid >> 5;
  const int c0  = tid * 8;

  const float* kr = K + (size_t)n * HDIM;
  const float* vr = V + (size_t)n * HDIM;
  const float* hr = hidden + (size_t)n * HDIM;

  float h[8], kk[8], vv[8];
  *(float4*)&h[0]  = *(const float4*)(hr + c0);
  *(float4*)&h[4]  = *(const float4*)(hr + c0 + 4);
  *(float4*)&kk[0] = *(const float4*)(kr + c0);
  *(float4*)&kk[4] = *(const float4*)(kr + c0 + 4);
  *(float4*)&vv[0] = *(const float4*)(vr + c0);
  *(float4*)&vv[4] = *(const float4*)(vr + c0 + 4);

  float sh = 0.f, sk = 0.f, sv = 0.f, sd = 0.f;
#pragma unroll
  for (int j = 0; j < 8; ++j) {
    const int c = c0 + j;
    sh += h[j] * h[j];
    sk += kk[j] * kk[j];
    sv += vv[j] * vv[j];
    sd += h[j] * qn[c] * kk[j] * kn[c];
  }

#pragma unroll
  for (int off = 16; off > 0; off >>= 1) {
    sh += __shfl_xor(sh, off, 32);
    sk += __shfl_xor(sk, off, 32);
    sv += __shfl_xor(sv, off, 32);
    sd += __shfl_xor(sd, off, 32);
  }
  if (lane == 0) {
    red[wid * 4 + 0] = sh;
    red[wid * 4 + 1] = sk;
    red[wid * 4 + 2] = sv;
    red[wid * 4 + 3] = sd;
  }
  __syncthreads();
  float th = 0.f, tk = 0.f, tv = 0.f, td = 0.f;
#pragma unroll
  for (int i = 0; i < 8; ++i) {
    th += red[i * 4 + 0];
    tk += red[i * 4 + 1];
    tv += red[i * 4 + 2];
    td += red[i * 4 + 3];
  }

  const float inv_h = 1.0f / (float)HDIM;
  const float rms_h = sqrtf(th * inv_h + 1e-8f);
  const float rms_k = sqrtf(tk * inv_h + 1e-8f);
  const float rms_v = sqrtf(tv * inv_h + 1e-8f);
  const float logit = td / (rms_h * rms_k) * 0.02209708691207961f; // 1/sqrt(2048)
  const float gate  = 1.0f / (1.0f + expf(-logit));
  const float rinv_v = 1.0f / rms_v;

  float o[8];
#pragma unroll
  for (int j = 0; j < 8; ++j) {
    const int c = c0 + j;
    const float vno   = vv[j] * rinv_v * vn[c];
    const float gated = gate * vno;
    const float conv  = gated * cw[3 * c + 2] + cb[c];
    const float silu  = conv / (1.0f + expf(-conv));
    o[j] = silu + gated;
  }
  float* orow = out + (size_t)n * HDIM;
  *(float4*)(orow + c0)     = *(const float4*)&o[0];
  *(float4*)(orow + c0 + 4) = *(const float4*)&o[4];
}

// ---------------------------------------------------------------------------
extern "C" void kernel_launch(void* const* d_in, const int* in_sizes, int n_in,
                              void* d_out, int out_size, void* d_ws, size_t ws_size,
                              hipStream_t stream) {
  const float* hidden = (const float*)d_in[0];
  const int*   ids    = (const int*)  d_in[1];
  const float* tables = (const float*)d_in[2];
  const float* key_w  = (const float*)d_in[3];
  const float* val_w  = (const float*)d_in[4];
  const float* qn_w   = (const float*)d_in[5];
  const float* kn_w   = (const float*)d_in[6];
  const float* vn_w   = (const float*)d_in[7];
  const float* conv_w = (const float*)d_in[8];
  const float* conv_b = (const float*)d_in[9];
  float* out = (float*)d_out;

  // workspace layout
  unsigned short* memB = (unsigned short*)d_ws;                  // 8192*2048 bf16
  unsigned short* kB   = memB + (size_t)N_TOK * MDIM;            // 2048*2048 bf16
  unsigned short* vB   = kB + (size_t)HDIM * MDIM;               // 2048*2048 bf16
  float* Kf = (float*)(vB + (size_t)HDIM * MDIM);                // 8192*2048 f32
  float* Vf = Kf + (size_t)N_TOK * HDIM;                         // 8192*2048 f32

  gather_bf16_kernel<<<N_TOK, 256, 0, stream>>>(tables, ids, memB);

  cvt_w_kernel<<<(HDIM * MDIM / 4) / 256, 256, 0, stream>>>(key_w, val_w, kB, vB);

  dim3 gemmGrid(HDIM / 64, N_TOK / 128);
  dual_gemm_bf16_kernel<<<gemmGrid, 256, 0, stream>>>(memB, kB, vB, Kf, Vf);

  epilogue_kernel<<<N_TOK, 256, 0, stream>>>(Kf, Vf, hidden, qn_w, kn_w, vn_w,
                                             conv_w, conv_b, out);
  (void)in_sizes; (void)n_in; (void)out_size; (void)ws_size;
}
